// CustomLSTM_28664611733559
// MI455X (gfx1250) — matrix operand order repaired and checked by reading, not compiled, not run on hardware
//
#include <hip/hip_runtime.h>
#include <stdint.h>
#include <stddef.h>

// ---------------------------------------------------------------------------
// Persistent 2-layer LSTM for MI455X (gfx1250, wave32, WMMA bf16 16x16x32).
//
// Grid: 128 persistent workgroups = 8 batch-groups (M-tile 16) x 16 WGs that
// partition the 1024 hidden columns. 4 waves/WG; each wave owns one 16-col
// j-tile and accumulates the 4 gate N-tiles for it (4 x v8f accumulators).
// Weights are prepacked into per-lane WMMA B-fragment order in the workspace
// so the inner loop is: 2x ds_load_b128 (A from LDS) + 8x global_load_b128
// (B streamed from L2) + 4x v_wmma_f32_16x16x32_bf16, branch-free so the
// compiler can software-pipeline with partial s_wait_loadcnt.
// (No explicit prefetch: gfx1250 __builtin_prefetch only reaches SE/SYS scope,
//  which per the ISA never prefetches into the WGP — weights are already
//  L2-resident, so such prefetches are pure overhead and they add control
//  flow that defeats the load/WMMA pipelining.)
// ---------------------------------------------------------------------------

typedef __attribute__((ext_vector_type(16))) __bf16 v16bf;
typedef __attribute__((ext_vector_type(8)))  float  v8f;

#define HDIM    1024
#define BATCH   128
#define TSTEPS  512
#define INDIM   64
#define K0      (INDIM + HDIM)   // 1088 : layer-0 K  = [x_t | h0]
#define KT0     (K0 / 32)        // 34
#define K1      (2 * HDIM)       // 2048 : layer-1 K  = [h0' | h1]
#define KT1     (K1 / 32)        // 64
#define NGATE   (4 * HDIM)       // 4096 gate columns (i,f,o,g)
#define GROUPS  8                // batch groups of 16 (WMMA M)
#define WPG     16               // workgroups per group (j-slice of 64)
#define NBLOCKS (GROUPS * WPG)   // 128 persistent workgroups
#define NTHREADS 128             // 4 waves
#define ASTRIDE (K1 + 8)         // LDS row stride (bf16 elems), +8 kills 4KB-stride bank conflicts

union Frag {
    uint4 u[2];
    v16bf v;
};

__device__ __forceinline__ unsigned short f2bf(float f) {
    unsigned u = __float_as_uint(f);
    unsigned r = u + 0x7FFFu + ((u >> 16) & 1u);   // round-to-nearest-even
    return (unsigned short)(r >> 16);
}
__device__ __forceinline__ float bf2f(unsigned short b) {
    return __uint_as_float(((unsigned)b) << 16);
}

__device__ __forceinline__ v8f splat8(float x) {
    v8f r;
#pragma unroll
    for (int i = 0; i < 8; ++i) r[i] = x;
    return r;
}
__device__ __forceinline__ float sig1(float x) {
    return 1.0f / (1.0f + __expf(-x));
}
__device__ __forceinline__ float tanh1(float x) {
    x = fminf(15.0f, fmaxf(-15.0f, x));
    float e = __expf(-2.0f * x);
    return (1.0f - e) / (1.0f + e);
}
__device__ __forceinline__ v8f sigv(v8f x) {
    v8f r;
#pragma unroll
    for (int i = 0; i < 8; ++i) r[i] = sig1(x[i]);
    return r;
}
__device__ __forceinline__ v8f tanhv(v8f x) {
    v8f r;
#pragma unroll
    for (int i = 0; i < 8; ++i) r[i] = tanh1(x[i]);
    return r;
}

__device__ __forceinline__ v8f wmma_bf16(v16bf a, v16bf b, v8f c) {
    // v_wmma_f32_16x16x32_bf16  (neg_a, A, neg_b, B, c_mod, C, reuse_a, reuse_b)
    return __builtin_amdgcn_wmma_f32_16x16x32_bf16(false, a, false, b, (short)0, c,
                                                   false, false);
}

// Per-group monotonic barrier over global atomics (16 arrivals each).
__device__ __forceinline__ void groupbar(unsigned int* cnt, unsigned int target) {
    __threadfence();            // release our h-stores to device scope
    __syncthreads();
    if (threadIdx.x == 0) {
        __hip_atomic_fetch_add(cnt, 1u, __ATOMIC_RELEASE, __HIP_MEMORY_SCOPE_AGENT);
        while (__hip_atomic_load(cnt, __ATOMIC_ACQUIRE, __HIP_MEMORY_SCOPE_AGENT) < target)
            __builtin_amdgcn_s_sleep(1);
    }
    __syncthreads();
}

// ---------------------------------------------------------------------------
// Prologue kernels
// ---------------------------------------------------------------------------

// Pack [Wx|Uh] (fp32, row-major [4096, K]) into bf16 B-fragment order:
//   elem index = ((nt*KT + kt)*32 + lane)*16 + e  holds  W[n = nt*16+lane%16, k = kt*32+16*(e/8)+8*(lane/16)+e%8]
// so each lane of a wave loads its 32B fragment with two global_load_b128.
__global__ void pack_w_kernel(const float* __restrict__ Wx0, const float* __restrict__ Uh0,
                              const float* __restrict__ Wx1, const float* __restrict__ Uh1,
                              unsigned short* __restrict__ W0p, unsigned short* __restrict__ W1p) {
    const long n0 = (long)NGATE * K0;
    const long n1 = (long)NGATE * K1;
    long idx = (long)blockIdx.x * blockDim.x + threadIdx.x;
    if (idx >= n0 + n1) return;
    if (idx < n0) {
        int  e  = (int)(idx & 15);
        long t  = idx >> 4;
        int  ln = (int)(t & 31); t >>= 5;
        int  kt = (int)(t % KT0);
        int  nt = (int)(t / KT0);
        int  n = nt * 16 + (ln & 15), k = kt * 32 + ((e & 8) << 1) + (ln >> 4) * 8 + (e & 7);
        float v = (k < INDIM) ? Wx0[(long)n * INDIM + k]
                              : Uh0[(long)n * HDIM + (k - INDIM)];
        W0p[idx] = f2bf(v);
    } else {
        long j  = idx - n0;
        int  e  = (int)(j & 15);
        long t  = j >> 4;
        int  ln = (int)(t & 31); t >>= 5;
        int  kt = (int)(t % KT1);
        int  nt = (int)(t / KT1);
        int  n = nt * 16 + (ln & 15), k = kt * 32 + ((e & 8) << 1) + (ln >> 4) * 8 + (e & 7);
        float v = (k < HDIM) ? Wx1[(long)n * HDIM + k]
                             : Uh1[(long)n * HDIM + (k - HDIM)];
        W1p[j] = f2bf(v);
    }
}

// x [B,T,64] fp32 -> xbf [T,B,64] bf16 (time-major so each step reads contiguously)
__global__ void pack_x_kernel(const float* __restrict__ x, unsigned short* __restrict__ xbf) {
    long idx = (long)blockIdx.x * blockDim.x + threadIdx.x;
    if (idx >= (long)TSTEPS * BATCH * INDIM) return;
    int k = (int)(idx % INDIM);
    int b = (int)((idx / INDIM) % BATCH);
    int t = (int)(idx / ((long)INDIM * BATCH));
    xbf[idx] = f2bf(x[((long)b * TSTEPS + t) * INDIM + k]);
}

// bias{0,1} = bx + bh ;  bi{0,1}[b,j] = boundary[b,:] @ Wb[j,:] + bb[j]
__global__ void pack_misc_kernel(const float* __restrict__ bx0, const float* __restrict__ bh0,
                                 const float* __restrict__ bx1, const float* __restrict__ bh1,
                                 const float* __restrict__ boundary,
                                 const float* __restrict__ Wb0, const float* __restrict__ bb0,
                                 const float* __restrict__ Wb1, const float* __restrict__ bb1,
                                 float* __restrict__ bias0, float* __restrict__ bias1,
                                 float* __restrict__ bi0, float* __restrict__ bi1) {
    int idx = blockIdx.x * blockDim.x + threadIdx.x;
    if (idx < NGATE) {
        bias0[idx] = bx0[idx] + bh0[idx];
    } else if (idx < 2 * NGATE) {
        int i = idx - NGATE;
        bias1[i] = bx1[i] + bh1[i];
    } else if (idx < 2 * NGATE + BATCH * HDIM) {
        int i = idx - 2 * NGATE;
        int j = i % HDIM, b = i / HDIM;
        bi0[i] = boundary[b * 2 + 0] * Wb0[j * 2 + 0] +
                 boundary[b * 2 + 1] * Wb0[j * 2 + 1] + bb0[j];
    } else if (idx < 2 * NGATE + 2 * BATCH * HDIM) {
        int i = idx - 2 * NGATE - BATCH * HDIM;
        int j = i % HDIM, b = i / HDIM;
        bi1[i] = boundary[b * 2 + 0] * Wb1[j * 2 + 0] +
                 boundary[b * 2 + 1] * Wb1[j * 2 + 1] + bb1[j];
    }
}

// zero h double-buffers and barrier counters (must happen every call)
__global__ void zero_kernel(unsigned short* __restrict__ h0buf,
                            unsigned short* __restrict__ h1buf,
                            unsigned int* __restrict__ barcnt) {
    int idx = blockIdx.x * blockDim.x + threadIdx.x;
    if (idx < 2 * BATCH * HDIM) { h0buf[idx] = 0; h1buf[idx] = 0; }
    if (idx < 512) barcnt[idx] = 0;
}

// ---------------------------------------------------------------------------
// Persistent LSTM kernel
// ---------------------------------------------------------------------------
__global__ void __launch_bounds__(NTHREADS, 1)
lstm_persist(const uint4* __restrict__ W0p, const uint4* __restrict__ W1p,
             const unsigned short* __restrict__ xbf,
             const float* __restrict__ bi0, const float* __restrict__ bi1,
             const float* __restrict__ bias0, const float* __restrict__ bias1,
             unsigned short* __restrict__ h0buf,   // [2][B][H] bf16 (double-buffered)
             unsigned short* __restrict__ h1buf,   // [2][B][H]
             unsigned int* __restrict__ barcnt) {
    extern __shared__ unsigned short lds[];        // [16][ASTRIDE] activation tile

    const int tid   = threadIdx.x;
    const int group = blockIdx.x / WPG;            // batch group (M-tile base)
    const int wgn   = blockIdx.x % WPG;            // j-slice within group
    const int Mbase = group * 16;
    const int wave  = tid >> 5;
    const int lane  = tid & 31;
    const int lhalf = lane >> 4;                   // 0/1 : which half-lane group
    const int nlo   = lane & 15;                   // N (and C/D column) within tile
    const int mrow  = lane & 15;                   // A-fragment row
    const int jtile = wgn * 64 + wave * 16;        // 16 hidden cols owned by this wave
    unsigned int* cnt = barcnt + group * 64;       // 256B-separated counters

    // Time-invariant per-lane data in C/D layout: M = r + 8*lhalf, N = nlo.
    v8f bif0, bif1, c0, c1;
    float bv0[4], bv1[4];
#pragma unroll
    for (int r = 0; r < 8; ++r) {
        int b = Mbase + r + lhalf * 8;
        bif0[r] = bi0[b * HDIM + jtile + nlo];
        bif1[r] = bi1[b * HDIM + jtile + nlo];
        c0[r] = 0.0f; c1[r] = 0.0f;
    }
    const uint4* bp0[4];
    const uint4* bp1[4];
#pragma unroll
    for (int g = 0; g < 4; ++g) {
        bv0[g] = bias0[g * HDIM + jtile + nlo];
        bv1[g] = bias1[g * HDIM + jtile + nlo];
        int nt = (g * HDIM + jtile) >> 4;          // global N-tile index
        bp0[g] = W0p + (size_t)nt * KT0 * 64 + lane * 2;   // uint4 units
        bp1[g] = W1p + (size_t)nt * KT1 * 64 + lane * 2;
    }

    const unsigned short* arow = &lds[mrow * ASTRIDE + lhalf * 8];

#pragma unroll 1
    for (int s = 0; s < TSTEPS; ++s) {
        const int rd = (s + 1) & 1;                // state after step s-1
        const int wr = s & 1;                      // state after step s

        // ---- layer 0: fill LDS with [x_t | h0_old], 16 rows x 1088 cols
        {
            const uint4* xs = (const uint4*)(xbf + ((size_t)s * BATCH + Mbase) * INDIM);
            uint4 xv = xs[tid];                    // 16*64 bf16 = 128 uint4
            *(uint4*)&lds[(tid >> 3) * ASTRIDE + (tid & 7) * 8] = xv;
            const uint4* hs = (const uint4*)(h0buf + ((size_t)rd * BATCH + Mbase) * HDIM);
#pragma unroll 4
            for (int i = tid; i < 16 * HDIM / 8; i += NTHREADS)
                *(uint4*)&lds[(i >> 7) * ASTRIDE + INDIM + (i & 127) * 8] = hs[i];
        }
        __syncthreads();

        // ---- layer 0 GEMM: [16 x 1088] x [1088 x 16] per gate (branch-free body)
        v8f acc[4];
#pragma unroll
        for (int g = 0; g < 4; ++g) acc[g] = splat8(bv0[g]);
        for (int kt = 0; kt < KT0; ++kt) {
            Frag a;
            a.u[0] = *(const uint4*)(arow + kt * 32);
            a.u[1] = *(const uint4*)(arow + kt * 32 + 16);
#pragma unroll
            for (int g = 0; g < 4; ++g) {
                Frag b;
                b.u[0] = bp0[g][kt * 64 + 0];
                b.u[1] = bp0[g][kt * 64 + 1];
                acc[g] = wmma_bf16(a.v, b.v, acc[g]);
            }
        }
        // ---- layer 0 cell (all in C/D layout)
        {
            v8f ii = sigv(acc[0]);
            v8f ff = sigv(acc[1] + bif0);
            v8f oo = sigv(acc[2]);
            v8f gg = tanhv(acc[3]);
            c0 = ff * c0 + ii * gg;
            v8f h = oo * tanhv(c0);
            unsigned short* hd = h0buf + (size_t)wr * BATCH * HDIM;
#pragma unroll
            for (int r = 0; r < 8; ++r)
                hd[(Mbase + r + lhalf * 8) * HDIM + jtile + nlo] = f2bf(h[r]);
        }
        groupbar(cnt, (unsigned)(2 * s + 1) * WPG);

        // ---- layer 1: fill LDS with [h0_new | h1_old], 16 x 2048
        {
            const uint4* h0s = (const uint4*)(h0buf + ((size_t)wr * BATCH + Mbase) * HDIM);
            const uint4* h1s = (const uint4*)(h1buf + ((size_t)rd * BATCH + Mbase) * HDIM);
#pragma unroll 4
            for (int i = tid; i < 16 * HDIM / 8; i += NTHREADS) {
                int r = i >> 7, cc = (i & 127) * 8;
                *(uint4*)&lds[r * ASTRIDE + cc] = h0s[i];
                *(uint4*)&lds[r * ASTRIDE + HDIM + cc] = h1s[i];
            }
        }
        __syncthreads();

        // ---- layer 1 GEMM: [16 x 2048] x [2048 x 16] per gate (branch-free body)
#pragma unroll
        for (int g = 0; g < 4; ++g) acc[g] = splat8(bv1[g]);
        for (int kt = 0; kt < KT1; ++kt) {
            Frag a;
            a.u[0] = *(const uint4*)(arow + kt * 32);
            a.u[1] = *(const uint4*)(arow + kt * 32 + 16);
#pragma unroll
            for (int g = 0; g < 4; ++g) {
                Frag b;
                b.u[0] = bp1[g][kt * 64 + 0];
                b.u[1] = bp1[g][kt * 64 + 1];
                acc[g] = wmma_bf16(a.v, b.v, acc[g]);
            }
        }
        // ---- layer 1 cell
        {
            v8f ii = sigv(acc[0]);
            v8f ff = sigv(acc[1] + bif1);
            v8f oo = sigv(acc[2]);
            v8f gg = tanhv(acc[3]);
            c1 = ff * c1 + ii * gg;
            v8f h = oo * tanhv(c1);
            unsigned short* hd = h1buf + (size_t)wr * BATCH * HDIM;
#pragma unroll
            for (int r = 0; r < 8; ++r)
                hd[(Mbase + r + lhalf * 8) * HDIM + jtile + nlo] = f2bf(h[r]);
        }
        groupbar(cnt, (unsigned)(2 * s + 2) * WPG);
    }
}

// ---------------------------------------------------------------------------
// Output head: out[b] = h1_final[b,:] . fcW + fcb
// ---------------------------------------------------------------------------
__global__ void head_kernel(const unsigned short* __restrict__ h1,
                            const float* __restrict__ fcW, const float* __restrict__ fcb,
                            float* __restrict__ out) {
    __shared__ float red[128];
    int b = blockIdx.x;
    float s = 0.0f;
    for (int j = threadIdx.x; j < HDIM; j += 128)
        s += bf2f(h1[(size_t)b * HDIM + j]) * fcW[j];
    red[threadIdx.x] = s;
    __syncthreads();
    for (int k = 64; k > 0; k >>= 1) {
        if (threadIdx.x < k) red[threadIdx.x] += red[threadIdx.x + k];
        __syncthreads();
    }
    if (threadIdx.x == 0) out[b] = red[0] + fcb[0];
}

// ---------------------------------------------------------------------------
extern "C" void kernel_launch(void* const* d_in, const int* in_sizes, int n_in,
                              void* d_out, int out_size, void* d_ws, size_t ws_size,
                              hipStream_t stream) {
    const float* x        = (const float*)d_in[0];
    const float* boundary = (const float*)d_in[1];
    const float* Wx0      = (const float*)d_in[2];
    const float* bx0      = (const float*)d_in[3];
    const float* Uh0      = (const float*)d_in[4];
    const float* bh0      = (const float*)d_in[5];
    const float* Wb0      = (const float*)d_in[6];
    const float* bb0      = (const float*)d_in[7];
    const float* Wx1      = (const float*)d_in[8];
    const float* bx1      = (const float*)d_in[9];
    const float* Uh1      = (const float*)d_in[10];
    const float* bh1      = (const float*)d_in[11];
    const float* Wb1      = (const float*)d_in[12];
    const float* bb1      = (const float*)d_in[13];
    const float* fcW      = (const float*)d_in[14];
    const float* fcb      = (const float*)d_in[15];
    float* out = (float*)d_out;
    (void)in_sizes; (void)n_in; (void)out_size; (void)ws_size;

    // workspace carve-up (256B aligned), total ~36 MB
    char* p = (char*)d_ws;
    auto alloc = [&](size_t bytes) -> char* {
        char* r = p;
        p += (bytes + 255) & ~(size_t)255;
        return r;
    };
    unsigned short* W0p   = (unsigned short*)alloc((size_t)NGATE * K0 * 2);
    unsigned short* W1p   = (unsigned short*)alloc((size_t)NGATE * K1 * 2);
    unsigned short* xbf   = (unsigned short*)alloc((size_t)TSTEPS * BATCH * INDIM * 2);
    float*          bias0 = (float*)alloc((size_t)NGATE * 4);
    float*          bias1 = (float*)alloc((size_t)NGATE * 4);
    float*          bi0   = (float*)alloc((size_t)BATCH * HDIM * 4);
    float*          bi1   = (float*)alloc((size_t)BATCH * HDIM * 4);
    unsigned short* h0buf = (unsigned short*)alloc((size_t)2 * BATCH * HDIM * 2);
    unsigned short* h1buf = (unsigned short*)alloc((size_t)2 * BATCH * HDIM * 2);
    unsigned int*   barcnt = (unsigned int*)alloc(512 * 4);

    zero_kernel<<<(2 * BATCH * HDIM + 255) / 256, 256, 0, stream>>>(h0buf, h1buf, barcnt);

    long nw = (long)NGATE * K0 + (long)NGATE * K1;
    pack_w_kernel<<<(int)((nw + 255) / 256), 256, 0, stream>>>(Wx0, Uh0, Wx1, Uh1, W0p, W1p);

    long nx = (long)TSTEPS * BATCH * INDIM;
    pack_x_kernel<<<(int)((nx + 255) / 256), 256, 0, stream>>>(x, xbf);

    int nm = 2 * NGATE + 2 * BATCH * HDIM;
    pack_misc_kernel<<<(nm + 255) / 256, 256, 0, stream>>>(bx0, bh0, bx1, bh1, boundary,
                                                           Wb0, bb0, Wb1, bb1,
                                                           bias0, bias1, bi0, bi1);

    size_t shbytes = (size_t)16 * ASTRIDE * 2;   // 65792 B activation tile
    lstm_persist<<<NBLOCKS, NTHREADS, shbytes, stream>>>(
        (const uint4*)W0p, (const uint4*)W1p, xbf, bi0, bi1, bias0, bias1,
        h0buf, h1buf, barcnt);

    const unsigned short* h1fin = h1buf + (size_t)((TSTEPS - 1) & 1) * BATCH * HDIM;
    head_kernel<<<BATCH, 128, 0, stream>>>(h1fin, fcW, fcb, out);
}
